// ToroidalStateCore_63307817943080
// MI455X (gfx1250) — matrix operand order repaired and checked by reading, not compiled, not run on hardware
//
#include <hip/hip_runtime.h>
#include <math.h>

// ---------------------------------------------------------------------------
// ToroidalStateCore for MI455X (gfx1250, wave32, WMMA)
//   Pass 1: dual GEMM (phi & amp projections) in bf16 WMMA w/ f32 accum,
//           fused activations -> angle (tanh*pi + tesla gate) and amp (sigmoid)
//   Pass 2: sequential scan over T per (b,c) channel, sin/cos on the
//           transcendental pipe, fused clip + amp*state output.
// ---------------------------------------------------------------------------

typedef __bf16 v16bf __attribute__((ext_vector_type(16)));
typedef __bf16 v4bf  __attribute__((ext_vector_type(4)));
typedef float  v8f   __attribute__((ext_vector_type(8)));

#define B_ 8
#define T_ 2048
#define C_ 1024
#define M_ (B_ * T_)

#define TESLA_TOL 0.15f
#define TESLA_STEP 1.0471975511965976f      // pi/3
#define TESLA_INV_STEP 0.9549296585513720f  // 3/pi
#define PI_F 3.14159265358979323846f

// GEMM tiling: block computes 128(M) x 64(N) of BOTH outputs.
#define BM 128
#define BN 64
#define BK 32
#define SA_STRIDE 40  // 128x32 bf16 tile, +8 pad (80B rows, 16B aligned)
#define SW_STRIDE 40

union FragU {
    uint4 q[2];
    v16bf v;
};

// Fast device tanh: native v_tanh_f32 if the builtin exists, else one
// v_exp_f32 + v_rcp_f32 (tanh(x) = 1 - 2/(exp(2x)+1)).
__device__ __forceinline__ float fast_tanh(float x) {
#if __has_builtin(__builtin_amdgcn_tanhf)
    return __builtin_amdgcn_tanhf(x);
#else
    float e = __expf(2.0f * x);
    return 1.0f - 2.0f * __builtin_amdgcn_rcpf(e + 1.0f);
#endif
}

__device__ __forceinline__ float fast_sigmoid(float x) {
    return __builtin_amdgcn_rcpf(1.0f + __expf(-x));
}

__global__ __launch_bounds__(256) void dual_gemm_act_kernel(
    const float* __restrict__ x,   // [M_, C_]
    const float* __restrict__ Wp,  // [C_, C_] (row = output dim d, col = k)
    const float* __restrict__ bp,  // [C_]
    const float* __restrict__ Wa,  // [C_, C_]
    const float* __restrict__ ba,  // [C_]
    float* __restrict__ angle,     // [M_, C_] ws
    float* __restrict__ amp)       // [M_, C_] ws
{
    __shared__ __bf16 sA[2][BM * SA_STRIDE];
    __shared__ __bf16 sWp[2][BN * SW_STRIDE];
    __shared__ __bf16 sWa[2][BN * SW_STRIDE];

    const int tid  = threadIdx.x;
    const int lane = tid & 31;
    const int w    = tid >> 5;   // 0..7 (8 waves)
    const int mat  = w >> 2;     // 0 = phi, 1 = amp
    const int wq   = w & 3;
    const int wm   = wq & 1;     // which 64-row half of the M tile
    const int wn   = wq >> 1;    // which 32-col half of the N tile

    const int m0 = blockIdx.y * BM;
    const int n0 = blockIdx.x * BN;

    const int lhalf = lane >> 4;   // 0 or 1
    const int lmod  = lane & 15;

    // Stage one K-slice (f32 global -> bf16 LDS). 256 threads cooperate.
    auto stage = [&](int buf, int kk) {
        // X tile: 128 rows x 32 cols  (1024 float4 loads / 256 threads)
#pragma unroll
        for (int i = 0; i < 4; ++i) {
            int lin = tid + i * 256;
            int row = lin >> 3;
            int c4  = lin & 7;
            float4 v = *reinterpret_cast<const float4*>(
                x + (size_t)(m0 + row) * C_ + kk + c4 * 4);
            v4bf h;
            h[0] = (__bf16)v.x; h[1] = (__bf16)v.y;
            h[2] = (__bf16)v.z; h[3] = (__bf16)v.w;
            *reinterpret_cast<v4bf*>(&sA[buf][row * SA_STRIDE + c4 * 4]) = h;
        }
        // W tiles: 64 rows x 32 cols each (512 float4 loads / 256 threads)
#pragma unroll
        for (int i = 0; i < 2; ++i) {
            int lin = tid + i * 256;
            int row = lin >> 3;
            int c4  = lin & 7;
            float4 vp = *reinterpret_cast<const float4*>(
                Wp + (size_t)(n0 + row) * C_ + kk + c4 * 4);
            float4 va = *reinterpret_cast<const float4*>(
                Wa + (size_t)(n0 + row) * C_ + kk + c4 * 4);
            v4bf hp, ha;
            hp[0] = (__bf16)vp.x; hp[1] = (__bf16)vp.y;
            hp[2] = (__bf16)vp.z; hp[3] = (__bf16)vp.w;
            ha[0] = (__bf16)va.x; ha[1] = (__bf16)va.y;
            ha[2] = (__bf16)va.z; ha[3] = (__bf16)va.w;
            *reinterpret_cast<v4bf*>(&sWp[buf][row * SW_STRIDE + c4 * 4]) = hp;
            *reinterpret_cast<v4bf*>(&sWa[buf][row * SW_STRIDE + c4 * 4]) = ha;
        }
    };

    v8f acc[4][2] = {};

    stage(0, 0);
    __syncthreads();

    const int KSTEPS = C_ / BK;  // 32
    for (int ks = 0; ks < KSTEPS; ++ks) {
        const int buf = ks & 1;
        if (ks + 1 < KSTEPS) stage(buf ^ 1, (ks + 1) * BK);

        const __bf16* sAb = sA[buf];
        const __bf16* sWb = mat ? sWa[buf] : sWp[buf];

        // B fragments: 32(K)x16(N); lane n<16 -> col n, K 0..15;
        //              lane n+16  -> col n, K 16..31   (2x ds_load_b128 each)
        FragU bfr[2];
#pragma unroll
        for (int ni = 0; ni < 2; ++ni) {
            int r  = wn * 32 + ni * 16 + lmod;
            int kb = lhalf * 16;
            const __bf16* p = sWb + r * SW_STRIDE + kb;
            bfr[ni].q[0] = *reinterpret_cast<const uint4*>(p);
            bfr[ni].q[1] = *reinterpret_cast<const uint4*>(p + 8);
        }

        // A fragments: 16(M)x32(K); lane m<16 -> row m, K 0..15;
        //              lane m+16  -> row m, K 16..31
#pragma unroll
        for (int mi = 0; mi < 4; ++mi) {
            int rowl = wm * 64 + mi * 16 + lmod;
            int klo  = lhalf * 16;
            const __bf16* p = sAb + rowl * SA_STRIDE + klo;
            FragU afr;
            afr.q[0] = *reinterpret_cast<const uint4*>(p);
            afr.q[1] = *reinterpret_cast<const uint4*>(p + 8);
#pragma unroll
            for (int ni = 0; ni < 2; ++ni) {
                acc[mi][ni] = __builtin_amdgcn_wmma_f32_16x16x32_bf16(
                    false, afr.v, false, bfr[ni].v,
                    (short)0, acc[mi][ni], false, false);
            }
        }
        __syncthreads();
    }

    // Epilogue: bias + activation, write f32 intermediates.
    const float* bias = mat ? ba : bp;
    float* dst        = mat ? amp : angle;
#pragma unroll
    for (int ni = 0; ni < 2; ++ni) {
        const int colg = n0 + wn * 32 + ni * 16 + lmod;
        const float bv = bias[colg];
#pragma unroll
        for (int mi = 0; mi < 4; ++mi) {
#pragma unroll
            for (int r = 0; r < 8; ++r) {
                const int rowg = m0 + wm * 64 + mi * 16 + r + lhalf * 8;
                float v = acc[mi][ni][r] + bv;
                if (mat == 0) {
                    // angle = tesla_gate(tanh(.)*pi); rintf == RNE == jnp.round
                    float a  = fast_tanh(v) * PI_F;
                    float nr = rintf(a * TESLA_INV_STEP) * TESLA_STEP;
                    v = (fabsf(a - nr) < TESLA_TOL) ? nr : a;
                } else {
                    v = fast_sigmoid(v);
                }
                dst[(size_t)rowg * C_ + colg] = v;
            }
        }
    }
}

// ---------------------------------------------------------------------------
// Sequential scan over T. One thread per (b, c) channel (8192 chains).
// Register prefetch of next step + global_prefetch 8 steps ahead to hide
// latency on the dependent recurrence. Last iteration peeled so the hot
// loop is branch-free.
// ---------------------------------------------------------------------------
__global__ __launch_bounds__(128) void toroidal_scan_kernel(
    const float* __restrict__ angle, const float* __restrict__ amp,
    const float* __restrict__ state_in, float* __restrict__ out,
    float* __restrict__ state_out)
{
    const int idx = blockIdx.x * 128 + threadIdx.x;  // 0 .. B_*C_-1
    const int b = idx >> 10;        // / C_
    const int c = idx & (C_ - 1);

    float s = state_in[idx];
    const size_t base = (size_t)b * T_ * C_ + c;

    float aN = angle[base];
    float mN = amp[base];

    auto step = [&](float a, float m, int t) {
        float sn, cs;
        __sincosf(a, &sn, &cs);
        s = __builtin_fmaf(cs, s, -sn * (1.0f - s));
        s = fminf(1.0f, fmaxf(-1.0f, s));
        out[base + (size_t)t * C_] = m * s;
    };

    for (int t = 0; t < T_ - 1; ++t) {
        const float a = aN;
        const float m = mN;
        const size_t nb = base + (size_t)(t + 1) * C_;
        aN = angle[nb];
        mN = amp[nb];
        if (t + 8 < T_) {
            const size_t pb = base + (size_t)(t + 8) * C_;
            __builtin_prefetch(angle + pb, 0, 0);
            __builtin_prefetch(amp + pb, 0, 0);
        }
        step(a, m, t);
    }
    step(aN, mN, T_ - 1);

    state_out[idx] = s;
}

// ---------------------------------------------------------------------------
extern "C" void kernel_launch(void* const* d_in, const int* in_sizes, int n_in,
                              void* d_out, int out_size, void* d_ws, size_t ws_size,
                              hipStream_t stream) {
    (void)in_sizes; (void)n_in; (void)out_size; (void)ws_size;

    const float* x   = (const float*)d_in[0];  // [B,T,C]
    const float* st  = (const float*)d_in[1];  // [B,C]
    const float* Wp  = (const float*)d_in[2];  // [C,C]
    const float* bp  = (const float*)d_in[3];  // [C]
    const float* Wa  = (const float*)d_in[4];  // [C,C]
    const float* ba  = (const float*)d_in[5];  // [C]

    float* out       = (float*)d_out;                  // [B,T,C]
    float* state_out = out + (size_t)M_ * C_;          // [B,C] (tuple concat)

    float* angle_ws  = (float*)d_ws;                   // 64 MB
    float* amp_ws    = angle_ws + (size_t)M_ * C_;     // 64 MB

    dim3 grid1(C_ / BN, M_ / BM);  // (16, 128)
    dual_gemm_act_kernel<<<grid1, 256, 0, stream>>>(
        x, Wp, bp, Wa, ba, angle_ws, amp_ws);

    toroidal_scan_kernel<<<(B_ * C_) / 128, 128, 0, stream>>>(
        angle_ws, amp_ws, st, out, state_out);
}
